// ManualMultiHeadSelfAttention_22359599743517
// MI455X (gfx1250) — compile-verified, matrix operand order checked by reading, not run on hardware
//
#include <hip/hip_runtime.h>
#include <hip/hip_bf16.h>

// ---------------------------------------------------------------------------
// Manual multi-head self-attention for gfx1250 (MI455X), bf16 WMMA path.
// D=768, H=12, Dh=64, B=2, N=4096. All matmuls via v_wmma_f32_16x16x32_bf16.
// GEMM waves compute 16x128 strips: A-fragment reused across 8 WMMAs/k-step.
// ---------------------------------------------------------------------------

#define D_MODEL 768
#define N_HEADS 12
#define HEAD_DIM 64
#define SEQ 4096
#define BATCH 2
#define M_TOKENS (BATCH * SEQ)   // 8192
#define QKV_N (3 * D_MODEL)      // 2304

typedef __bf16 bf16;
typedef __attribute__((ext_vector_type(16))) __bf16 v16bf;
typedef __attribute__((ext_vector_type(8))) float v8f;

__device__ __forceinline__ void ldpair(const bf16* p, bf16& a, bf16& b) {
  union {
    unsigned u;
    bf16 h[2];
  } c;
  c.u = *reinterpret_cast<const unsigned*>(p);  // 4-byte aligned pair load
  a = c.h[0];
  b = c.h[1];
}

// Load a 16x32 bf16 WMMA fragment (A-fragment from row-major [m][k] data, or
// B-fragment from [n][k]-major data). ISA 7.12.2 16-bit A layout:
//   lanes 0-15 : VGPR0-3 -> K=0..7 (pairs), VGPR4-7 -> K=16..23
//   lanes 16-31: VGPR0-3 -> K=8..15,        VGPR4-7 -> K=24..31
// Per lane this is two contiguous 16B regions -> 2x global_load_b128.
__device__ __forceinline__ v16bf load_frag(const bf16* src, int row0, int k0,
                                           int ld) {
  const int lane = threadIdx.x & 31;
  const int lo = lane & 15;
  const int hi = (lane >> 4) & 1;
  const bf16* p = src + (size_t)(row0 + lo) * ld + k0 + hi * 8;
  v16bf f;
#pragma unroll
  for (int r = 0; r < 8; ++r) {
    const int kk = (r < 4) ? (2 * r) : (16 + 2 * (r - 4));
    bf16 x, y;
    ldpair(p + kk, x, y);
    f[2 * r] = x;
    f[2 * r + 1] = y;
  }
  return f;
}

__device__ __forceinline__ v8f wmma_bf16(v16bf a, v16bf b, v8f c) {
  // D = A(16x32 bf16) * B(32x16 bf16) + C(16x16 f32)
  return __builtin_amdgcn_wmma_f32_16x16x32_bf16(false, a, false, b, (short)0,
                                                 c, false, false);
}

// ---------------------------------------------------------------------------
// Precision-conversion helpers
// ---------------------------------------------------------------------------
__global__ void cvt_f32_bf16(const float* __restrict__ in,
                             bf16* __restrict__ out, int n) {
  int i = blockIdx.x * blockDim.x + threadIdx.x;
  if (i < n) out[i] = (bf16)in[i];
}

// in: [rows][cols] (k-major weights W[k][n]); out: [cols][rows] = W^T[n][k]
__global__ void transpose_f32_bf16(const float* __restrict__ in,
                                   bf16* __restrict__ out, int rows, int cols) {
  int i = blockIdx.x * blockDim.x + threadIdx.x;
  if (i < rows * cols) {
    int n = i / rows;
    int k = i % rows;
    out[(size_t)n * rows + k] = (bf16)in[(size_t)k * cols + n];
  }
}

// ---------------------------------------------------------------------------
// QKV projection: [8192 x 768] @ [768 x 2304] + bias. Each wave: 16 rows x
// 128 cols (8 WMMA tiles, A-fragment reused 8x per k-step). Epilogue scatters
// into Q[bh][tok][dh] (pre-scaled by 1/sqrt(Dh)), K[bh][tok][dh],
// V^T[bh][dh][tok].
// ---------------------------------------------------------------------------
__global__ __launch_bounds__(128) void qkv_gemm(
    const bf16* __restrict__ Xb, const bf16* __restrict__ Wqt,
    const float* __restrict__ bias, bf16* __restrict__ Qb,
    bf16* __restrict__ Kb, bf16* __restrict__ Vt) {
  const int wave = threadIdx.x >> 5;
  const int lane = threadIdx.x & 31;
  const int lo = lane & 15, hi = (lane >> 4) & 1;
  const int row0 = blockIdx.x * 64 + wave * 16;
  const int col0 = blockIdx.y * 128;

  v8f acc[8] = {{}, {}, {}, {}, {}, {}, {}, {}};
#pragma unroll 1
  for (int k0 = 0; k0 < D_MODEL; k0 += 32) {
    v16bf a = load_frag(Xb, row0, k0, D_MODEL);
#pragma unroll
    for (int t = 0; t < 8; ++t) {
      v16bf b = load_frag(Wqt, col0 + 16 * t, k0, D_MODEL);
      acc[t] = wmma_bf16(a, b, acc[t]);
    }
  }

#pragma unroll
  for (int t = 0; t < 8; ++t) {
    const int n = col0 + 16 * t + lo;
    const float bv = bias[n];
    const int which = n / D_MODEL;
    const int c = n % D_MODEL;
    const int h = c / HEAD_DIM;
    const int dh = c % HEAD_DIM;
#pragma unroll
    for (int r = 0; r < 8; ++r) {
      const int m = row0 + r + 8 * hi;    // global token row 0..8191
      const int b_ = m >> 12;             // batch
      const int tok = m & (SEQ - 1);
      const float v = acc[t][r] + bv;
      if (which == 0) {
        Qb[(((size_t)(b_ * N_HEADS + h)) * SEQ + tok) * HEAD_DIM + dh] =
            (bf16)(v * 0.125f);  // fold 1/sqrt(64) into Q
      } else if (which == 1) {
        Kb[(((size_t)(b_ * N_HEADS + h)) * SEQ + tok) * HEAD_DIM + dh] =
            (bf16)v;
      } else {
        Vt[(((size_t)(b_ * N_HEADS + h)) * HEAD_DIM + dh) * SEQ + tok] =
            (bf16)v;
      }
    }
  }
}

// ---------------------------------------------------------------------------
// Flash attention: one wave owns 16 query rows; streams 64-key blocks.
// S = Q K^T via WMMA, fp32 online softmax in the C-tile layout, P re-laid out
// through a per-wave LDS slice into A-fragments, O += P V via WMMA.
// ---------------------------------------------------------------------------
__global__ __launch_bounds__(128) void flash_attn(const bf16* __restrict__ Qb,
                                                  const bf16* __restrict__ Kb,
                                                  const bf16* __restrict__ Vt,
                                                  bf16* __restrict__ Ob) {
  __shared__ bf16 pbuf[4][16][64];  // per-wave 16x64 P tile, 8 KB
  const int wave = threadIdx.x >> 5;
  const int lane = threadIdx.x & 31;
  const int lo = lane & 15, hi = (lane >> 4) & 1;
  const int bh = blockIdx.x / (SEQ / 64);
  const int qblk = blockIdx.x % (SEQ / 64);
  const int b_ = bh / N_HEADS;
  const int h = bh % N_HEADS;
  const int q0 = qblk * 64 + wave * 16;

  const bf16* Qh = Qb + (size_t)bh * SEQ * HEAD_DIM;
  const bf16* Kh = Kb + (size_t)bh * SEQ * HEAD_DIM;
  const bf16* Vh = Vt + (size_t)bh * SEQ * HEAD_DIM;  // [dh][tok], ld = SEQ

  const v16bf qa0 = load_frag(Qh, q0, 0, HEAD_DIM);
  const v16bf qa1 = load_frag(Qh, q0, 32, HEAD_DIM);

  v8f o[4] = {{}, {}, {}, {}};
  float mrow[8], lrow[8];
#pragma unroll
  for (int r = 0; r < 8; ++r) {
    mrow[r] = -1e30f;
    lrow[r] = 0.f;
  }

#pragma unroll 1
  for (int j0 = 0; j0 < SEQ; j0 += 64) {
    if (j0 + 64 < SEQ) {  // global_prefetch_b8 next K tile
      __builtin_prefetch(Kh + (size_t)(j0 + 64) * HEAD_DIM + lane * 8, 0, 0);
    }
    // S tiles: 16 queries x 64 keys (4 col tiles)
    v8f s[4];
#pragma unroll
    for (int t = 0; t < 4; ++t) {
      v8f acc = {};
      v16bf kf = load_frag(Kh, j0 + 16 * t, 0, HEAD_DIM);
      acc = wmma_bf16(qa0, kf, acc);
      kf = load_frag(Kh, j0 + 16 * t, 32, HEAD_DIM);
      acc = wmma_bf16(qa1, kf, acc);
      s[t] = acc;
    }
    // Online softmax: row r lives in VGPR r across a 16-lane group.
#pragma unroll
    for (int r = 0; r < 8; ++r) {
      float x = fmaxf(fmaxf(s[0][r], s[1][r]), fmaxf(s[2][r], s[3][r]));
#pragma unroll
      for (int msk = 1; msk < 16; msk <<= 1)
        x = fmaxf(x, __shfl_xor(x, msk, 32));
      const float mnew = fmaxf(mrow[r], x);
      const float alpha = __expf(mrow[r] - mnew);
      mrow[r] = mnew;
      float rsum = 0.f;
#pragma unroll
      for (int t = 0; t < 4; ++t) {
        const float p = __expf(s[t][r] - mnew);
        s[t][r] = p;
        rsum += p;
        o[t][r] *= alpha;
      }
#pragma unroll
      for (int msk = 1; msk < 16; msk <<= 1)
        rsum += __shfl_xor(rsum, msk, 32);
      lrow[r] = lrow[r] * alpha + rsum;
    }
    // C-tile layout -> A-fragment layout through LDS (bf16).
#pragma unroll
    for (int t = 0; t < 4; ++t)
#pragma unroll
      for (int r = 0; r < 8; ++r)
        pbuf[wave][r + 8 * hi][16 * t + lo] = (bf16)s[t][r];
    __syncthreads();
    const v16bf pa0 = load_frag(&pbuf[wave][0][0], 0, 0, 64);
    const v16bf pa1 = load_frag(&pbuf[wave][0][0], 0, 32, 64);
#pragma unroll
    for (int t = 0; t < 4; ++t) {  // t = dh column tile
      v16bf vf = load_frag(Vh, 16 * t, j0, SEQ);
      o[t] = wmma_bf16(pa0, vf, o[t]);
      vf = load_frag(Vh, 16 * t, j0 + 32, SEQ);
      o[t] = wmma_bf16(pa1, vf, o[t]);
    }
    __syncthreads();
  }

  // Normalize and store O as bf16 in [tok][h*64+dh] (GEMM-ready layout).
#pragma unroll
  for (int r = 0; r < 8; ++r) {
    const float inv = 1.0f / lrow[r];
    const int q = q0 + r + 8 * hi;
    const size_t rowbase = ((size_t)b_ * SEQ + q) * D_MODEL + h * HEAD_DIM;
#pragma unroll
    for (int t = 0; t < 4; ++t)
      Ob[rowbase + 16 * t + lo] = (bf16)(o[t][r] * inv);
  }
}

// ---------------------------------------------------------------------------
// Output projection: [8192 x 768] @ [768 x 768] + bias -> fp32 d_out.
// Each wave: 16 rows x 128 cols (8 WMMA tiles per k-step).
// ---------------------------------------------------------------------------
__global__ __launch_bounds__(128) void out_gemm(const bf16* __restrict__ Ob,
                                                const bf16* __restrict__ Wot,
                                                const float* __restrict__ bias,
                                                float* __restrict__ out) {
  const int wave = threadIdx.x >> 5;
  const int lane = threadIdx.x & 31;
  const int lo = lane & 15, hi = (lane >> 4) & 1;
  const int row0 = blockIdx.x * 64 + wave * 16;
  const int col0 = blockIdx.y * 128;

  v8f acc[8] = {{}, {}, {}, {}, {}, {}, {}, {}};
#pragma unroll 1
  for (int k0 = 0; k0 < D_MODEL; k0 += 32) {
    v16bf a = load_frag(Ob, row0, k0, D_MODEL);
#pragma unroll
    for (int t = 0; t < 8; ++t) {
      v16bf b = load_frag(Wot, col0 + 16 * t, k0, D_MODEL);
      acc[t] = wmma_bf16(a, b, acc[t]);
    }
  }
#pragma unroll
  for (int t = 0; t < 8; ++t) {
    const int n = col0 + 16 * t + lo;
    const float bv = bias[n];
#pragma unroll
    for (int r = 0; r < 8; ++r) {
      const int m = row0 + r + 8 * hi;
      out[(size_t)m * D_MODEL + n] = acc[t][r] + bv;
    }
  }
}

// ---------------------------------------------------------------------------
extern "C" void kernel_launch(void* const* d_in, const int* in_sizes, int n_in,
                              void* d_out, int out_size, void* d_ws,
                              size_t ws_size, hipStream_t stream) {
  (void)in_sizes;
  (void)n_in;
  (void)out_size;
  (void)ws_size;
  const float* x = (const float*)d_in[0];
  const float* W_qkv = (const float*)d_in[1];
  const float* b_qkv = (const float*)d_in[2];
  const float* W_out = (const float*)d_in[3];
  const float* b_out = (const float*)d_in[4];
  float* out = (float*)d_out;

  char* ws = (char*)d_ws;
  size_t off = 0;
  auto alloc = [&](size_t bytes) -> void* {
    void* p = ws + off;
    off = (off + bytes + 255) & ~(size_t)255;
    return p;
  };
  const size_t headElems = (size_t)BATCH * N_HEADS * SEQ * HEAD_DIM;
  bf16* Xb = (bf16*)alloc((size_t)M_TOKENS * D_MODEL * sizeof(bf16));
  bf16* Wqt = (bf16*)alloc((size_t)QKV_N * D_MODEL * sizeof(bf16));
  bf16* Wot = (bf16*)alloc((size_t)D_MODEL * D_MODEL * sizeof(bf16));
  bf16* Qb = (bf16*)alloc(headElems * sizeof(bf16));
  bf16* Kb = (bf16*)alloc(headElems * sizeof(bf16));
  bf16* Vt = (bf16*)alloc(headElems * sizeof(bf16));
  bf16* Ob = (bf16*)alloc((size_t)M_TOKENS * D_MODEL * sizeof(bf16));

  {
    int n = M_TOKENS * D_MODEL;
    cvt_f32_bf16<<<(n + 255) / 256, 256, 0, stream>>>(x, Xb, n);
  }
  {
    int n = D_MODEL * QKV_N;
    transpose_f32_bf16<<<(n + 255) / 256, 256, 0, stream>>>(W_qkv, Wqt, D_MODEL,
                                                            QKV_N);
  }
  {
    int n = D_MODEL * D_MODEL;
    transpose_f32_bf16<<<(n + 255) / 256, 256, 0, stream>>>(W_out, Wot, D_MODEL,
                                                            D_MODEL);
  }
  qkv_gemm<<<dim3(M_TOKENS / 64, QKV_N / 128), 128, 0, stream>>>(Xb, Wqt,
                                                                 b_qkv, Qb, Kb,
                                                                 Vt);
  flash_attn<<<BATCH * N_HEADS * (SEQ / 64), 128, 0, stream>>>(Qb, Kb, Vt, Ob);
  out_gemm<<<dim3(M_TOKENS / 64, D_MODEL / 128), 128, 0, stream>>>(Ob, Wot,
                                                                   b_out, out);
}
